// VectorQuantizer_54030688584153
// MI455X (gfx1250) — matrix operand order repaired and checked by reading, not compiled, not run on hardware
//
#include <hip/hip_runtime.h>
#include <hip/hip_bf16.h>

typedef __attribute__((ext_vector_type(16))) __bf16 v16bf;
typedef __attribute__((ext_vector_type(8)))  float  v8f;
typedef __attribute__((ext_vector_type(4)))  unsigned int u32x4;
typedef __attribute__((ext_vector_type(8)))  int i32x8;
typedef __attribute__((ext_vector_type(4)))  int i32x4;

#define N_ROWS 32768
#define K_CODES 4096
#define D_DIM 512

// LDS layout (halves), all in WMMA fragment order, zero padding:
//  sAhi: 128 rows -> [w(8)][dIter(16)][lane(32)][16] = 65536 halves (128 KB)
//  sAlo: 65536 halves
//  sBhi: 32 codes -> [c(2)][dIter(16)][lane(32)][16] = 16384 halves (32 KB)
//  sBlo: 16384 halves  (contiguous after sBhi: one 64 KB TDM landing zone)
//  total = 327680 B = 320 KB (exact WGP LDS)
#define A_HALVES 65536
#define B_HALVES 16384
#define B_LDS_BYTE_OFF (2 * A_HALVES * 2)   // 262144: sBhi offset in dynamic LDS
#define TILE_BYTES 65536                     // combined hi|lo fragment tile

__device__ __forceinline__ unsigned short f2bf(float x) {
  unsigned u = __float_as_uint(x);
  u += 0x7FFFu + ((u >> 16) & 1u);          // round-to-nearest-even
  return (unsigned short)(u >> 16);
}
__device__ __forceinline__ float bf2f(unsigned short h) {
  return __uint_as_float(((unsigned)h) << 16);
}

// ---------------------------------------------------------------------------
// Kernel 1: split weight into bf16 hi/lo written in combined TILE-FRAGMENT
// order ([tile][hi 32KB | lo 32KB]), compute ||e_k||^2, zero loss accum.
//   fragOff = ((c*16 + i)*32 + half*16 + n)*16 + (rem>>4)*8 + (rem & 7)
//   where k = tile*32 + c*16 + n, d = i*32 + rem, half = (rem>>3)&1.
// ---------------------------------------------------------------------------
__global__ void vq_prep_kernel(const float* __restrict__ w,
                               unsigned short* __restrict__ wF,
                               float* __restrict__ wnorm,
                               float* __restrict__ acc) {
  __shared__ float ssum[8];
  const int k = blockIdx.x;
  const int t = threadIdx.x;
  const int tile = k >> 5;
  const int c = (k >> 4) & 1;
  const int n = k & 15;

  float2 w2 = ((const float2*)(w + (size_t)k * D_DIM))[t];
  unsigned short h0 = f2bf(w2.x), h1 = f2bf(w2.y);
  unsigned short l0 = f2bf(w2.x - bf2f(h0)), l1 = f2bf(w2.y - bf2f(h1));

  const int d = 2 * t;
  const int i = d >> 5;
  const int rem = d & 31;
  const int half = (rem >> 3) & 1;
  const int j0 = ((rem >> 4) << 3) + (rem & 7);          // even
  const unsigned fragOff = ((unsigned)((c * 16 + i) * 32 + half * 16 + n) << 4) + j0;
  const unsigned base = (unsigned)tile * 32768u;   // halves per combined tile
  *(unsigned int*)(wF + base + fragOff) = (unsigned)h0 | ((unsigned)h1 << 16);
  *(unsigned int*)(wF + base + 16384u + fragOff) = (unsigned)l0 | ((unsigned)l1 << 16);

  float sq = w2.x * w2.x + w2.y * w2.y;
  for (int offs = 16; offs >= 1; offs >>= 1) sq += __shfl_down(sq, offs, 32);
  if ((t & 31) == 0) ssum[t >> 5] = sq;
  __syncthreads();
  if (t == 0) {
    float s = 0.f;
    #pragma unroll
    for (int q = 0; q < 8; ++q) s += ssum[q];
    wnorm[k] = s;
    if (k == 0) *acc = 0.f;
  }
}

// ---------------------------------------------------------------------------
// Kernel 2: fused bf16x3 WMMA GEMM + argmin with TDM-staged B tiles.
// Block = 256 threads (8 waves); 128 input rows resident in LDS (fragment
// order). Per 32-code tile, wave 0 issues ONE tensor_load_to_lds DMA of the
// contiguous 64 KB fragment-order block onto sBhi|sBlo, waits TENSORcnt,
// then all 8 waves consume it.
// ---------------------------------------------------------------------------
__global__ void vq_argmin_kernel(const float* __restrict__ inp,
                                 const unsigned short* __restrict__ wF,
                                 const float* __restrict__ wnorm,
                                 int* __restrict__ idxOut,
                                 float* __restrict__ idxOutF) {
  extern __shared__ char smem_raw[];
  unsigned short* sAhi = (unsigned short*)smem_raw;
  unsigned short* sAlo = sAhi + A_HALVES;
  unsigned short* sBhi = sAlo + A_HALVES;
  unsigned short* sBlo = sBhi + B_HALVES;

  const int t = threadIdx.x;
  const int lane = t & 31;
  const int wv = t >> 5;
  const int m = lane & 15;
  const int half = lane >> 4;
  const int rowBase = blockIdx.x * 128;

  // ---- Fill A (hi/lo bf16) into LDS in fragment order ----
  const float4* gin = (const float4*)(inp + (size_t)rowBase * D_DIM);
  #pragma unroll 4
  for (int it = 0; it < 64; ++it) {
    int j = t + it * 256;            // float4 index 0..16383
    int row = j >> 7;
    int d = (j & 127) * 4;
    float4 x = gin[j];
    unsigned short h0 = f2bf(x.x), h1 = f2bf(x.y), h2 = f2bf(x.z), h3 = f2bf(x.w);
    unsigned short l0 = f2bf(x.x - bf2f(h0)), l1 = f2bf(x.y - bf2f(h1));
    unsigned short l2 = f2bf(x.z - bf2f(h2)), l3 = f2bf(x.w - bf2f(h3));
    int w8 = row >> 4, mm = row & 15;
    int i = d >> 5, rem = d & 31;
    int ah = (rem >> 3) & 1, chunk = rem >> 4, j0 = rem & 7;   // j0 in {0,4}
    unsigned off = ((unsigned)((w8 * 16 + i) * 32 + ah * 16 + mm) << 4) +
                   (unsigned)(chunk * 8 + j0);
    uint2 ph; ph.x = (unsigned)h0 | ((unsigned)h1 << 16); ph.y = (unsigned)h2 | ((unsigned)h3 << 16);
    uint2 pl; pl.x = (unsigned)l0 | ((unsigned)l1 << 16); pl.y = (unsigned)l2 | ((unsigned)l3 << 16);
    *(uint2*)(sAhi + off) = ph;
    *(uint2*)(sAlo + off) = pl;
  }

  float bv[8];
  int bi[8];
  #pragma unroll
  for (int r = 0; r < 8; ++r) { bv[r] = 3.4e38f; bi[r] = 0; }

  // TDM descriptor invariants (D# per CDNA5 ISA ch.8):
  // group1: data_size=8B [17:16]=3; tensor_dim0=8192 [79:48]; tensor_dim1=1
  // [111:80]; tile_dim0=8192 [127:112]; tile_dim1=1 [143:128];
  // tensor_dim0_stride=8192 [207:160]; tensor_dim1_stride=8192 [255:208].
  i32x8 g1;
  g1.s0 = 0x00030000;   // data_size = 3 (8 bytes)
  g1.s1 = 0x20000000;   // tensor_dim0[15:0]=8192 in bits [31:16]
  g1.s2 = 0x00010000;   // tensor_dim1[15:0]=1 in bits [31:16]
  g1.s3 = 0x20000000;   // tile_dim0=8192 in bits [31:16]
  g1.s4 = 0x00000001;   // tile_dim1=1, tile_dim2=0
  g1.s5 = 8192;         // tensor_dim0_stride[31:0]
  g1.s6 = 0x20000000;   // tensor_dim1_stride[15:0]=8192 in bits [31:16]
  g1.s7 = 0;
  i32x4 gz4 = {0, 0, 0, 0};
  i32x8 gz8 = {0, 0, 0, 0, 0, 0, 0, 0};
  const unsigned ldsB = (unsigned)__builtin_amdgcn_groupstaticsize() + (unsigned)B_LDS_BYTE_OFF;
  const unsigned long long wFaddr = (unsigned long long)(uintptr_t)wF;

  const char* wFbytes = (const char*)wF;

  for (int kt = 0; kt < K_CODES / 32; ++kt) {
    __syncthreads();  // previous tile's compute done before overwriting B
    if (wv == 0) {
      // ---- One 64 KB TDM copy: combined hi|lo fragment tile -> sBhi|sBlo ----
      unsigned long long ga = wFaddr + (unsigned long long)kt * TILE_BYTES;
      u32x4 g0;
      g0.x = 1u;                                        // count=1 valid descriptor
      g0.y = ldsB;                                      // lds_addr (bytes)
      g0.z = (unsigned)(ga & 0xFFFFFFFFull);            // global_addr[31:0]
      g0.w = (unsigned)((ga >> 32) & 0x01FFFFFFull) | (2u << 30);  // [56:32] | type=2
      __builtin_amdgcn_tensor_load_to_lds(g0, g1, gz4, gz4, gz8, 0);
      __builtin_amdgcn_s_wait_tensorcnt(0);
    }
    if (kt + 1 < K_CODES / 32)
      __builtin_prefetch((const void*)(wFbytes + (size_t)(kt + 1) * TILE_BYTES + t * 256), 0, 0);
    __syncthreads();

    // ---- bf16x3 WMMA over D for two 16-code column tiles ----
    v8f c0 = {0.f, 0.f, 0.f, 0.f, 0.f, 0.f, 0.f, 0.f};
    v8f c1 = {0.f, 0.f, 0.f, 0.f, 0.f, 0.f, 0.f, 0.f};
    #pragma unroll
    for (int i = 0; i < 16; ++i) {
      const unsigned aOff = (unsigned)((wv * 16 + i) * 32 + lane) << 4;
      v16bf ahv = *(const v16bf*)(sAhi + aOff);
      v16bf alv = *(const v16bf*)(sAlo + aOff);
      const unsigned b0Off = (unsigned)(i * 32 + lane) << 4;
      const unsigned b1Off = (unsigned)((16 + i) * 32 + lane) << 4;
      v16bf bh0 = *(const v16bf*)(sBhi + b0Off);
      v16bf bl0 = *(const v16bf*)(sBlo + b0Off);
      v16bf bh1 = *(const v16bf*)(sBhi + b1Off);
      v16bf bl1 = *(const v16bf*)(sBlo + b1Off);
      c0 = __builtin_amdgcn_wmma_f32_16x16x32_bf16(false, ahv, false, bh0, (short)0, c0, false, false);
      c0 = __builtin_amdgcn_wmma_f32_16x16x32_bf16(false, ahv, false, bl0, (short)0, c0, false, false);
      c0 = __builtin_amdgcn_wmma_f32_16x16x32_bf16(false, alv, false, bh0, (short)0, c0, false, false);
      c1 = __builtin_amdgcn_wmma_f32_16x16x32_bf16(false, ahv, false, bh1, (short)0, c1, false, false);
      c1 = __builtin_amdgcn_wmma_f32_16x16x32_bf16(false, ahv, false, bl1, (short)0, c1, false, false);
      c1 = __builtin_amdgcn_wmma_f32_16x16x32_bf16(false, alv, false, bh1, (short)0, c1, false, false);
    }

    // ---- Fused argmin epilogue: score = ||e||^2 - 2*dot ----
    const int k0 = kt * 32 + m;
    const int k1 = k0 + 16;
    const float wn0 = wnorm[k0];
    const float wn1 = wnorm[k1];
    #pragma unroll
    for (int r = 0; r < 8; ++r) {
      float s0 = wn0 - 2.0f * c0[r];
      if (s0 < bv[r]) { bv[r] = s0; bi[r] = k0; }
      float s1 = wn1 - 2.0f * c1[r];
      if (s1 < bv[r]) { bv[r] = s1; bi[r] = k1; }
    }
  }

  // ---- Cross-lane min+argmin within each 16-lane half ----
  #pragma unroll
  for (int off = 1; off < 16; off <<= 1) {
    #pragma unroll
    for (int r = 0; r < 8; ++r) {
      float ov = __shfl_xor(bv[r], off, 32);
      int oi = __shfl_xor(bi[r], off, 32);
      if (ov < bv[r] || (ov == bv[r] && oi < bi[r])) { bv[r] = ov; bi[r] = oi; }
    }
  }
  if (m == 0) {
    const int rbase = rowBase + wv * 16 + half * 8;   // lane0: rows +0..7, lane16: +8..15
    #pragma unroll
    for (int r = 0; r < 8; ++r) {
      idxOut[rbase + r] = bi[r];
      idxOutF[rbase + r] = (float)bi[r];
    }
  }
}

// ---------------------------------------------------------------------------
// Kernel 3: gather codebook rows (quantized_st == quantized numerically) and
// accumulate sum((q - x)^2). Block = 256 threads handles 2 rows (float4).
// ---------------------------------------------------------------------------
__global__ void vq_gather_loss_kernel(const float* __restrict__ inp,
                                      const float* __restrict__ w,
                                      const int* __restrict__ idxInt,
                                      float* __restrict__ outq,
                                      float* __restrict__ acc) {
  __shared__ float ssum[8];
  const int t = threadIdx.x;
  const int row = blockIdx.x * 2 + (t >> 7);
  const int dq = t & 127;
  const int idx = idxInt[row];
  float4 e = ((const float4*)w)[(size_t)idx * 128 + dq];
  float4 x = ((const float4*)inp)[(size_t)row * 128 + dq];
  float* o = outq + (size_t)row * D_DIM + dq * 4;   // outq = d_out+1: scalar stores
  o[0] = e.x; o[1] = e.y; o[2] = e.z; o[3] = e.w;
  float dx = e.x - x.x, dy = e.y - x.y, dz = e.z - x.z, dw = e.w - x.w;
  float sq = dx * dx + dy * dy + dz * dz + dw * dw;
  for (int off = 16; off >= 1; off >>= 1) sq += __shfl_down(sq, off, 32);
  if ((t & 31) == 0) ssum[t >> 5] = sq;
  __syncthreads();
  if (t == 0) {
    float s = 0.f;
    #pragma unroll
    for (int q = 0; q < 8; ++q) s += ssum[q];
    atomicAdd(acc, s);
  }
}

// vq_loss = q_latent + 0.25*e_latent = 1.25 * mean((q - x)^2)
__global__ void vq_finalize_kernel(const float* __restrict__ acc,
                                   float* __restrict__ out) {
  out[0] = acc[0] * (1.25f / ((float)N_ROWS * (float)D_DIM));
}

extern "C" void kernel_launch(void* const* d_in, const int* in_sizes, int n_in,
                              void* d_out, int out_size, void* d_ws, size_t ws_size,
                              hipStream_t stream) {
  const float* inputs = (const float*)d_in[0];   // [N, D] fp32
  const float* weight = (const float*)d_in[1];   // [K, D] fp32
  float* out = (float*)d_out;                    // [loss(1) | quantized(N*D) | indices(N)]

  // Workspace layout (~8.4 MB)
  char* ws = (char*)d_ws;
  unsigned short* wF = (unsigned short*)ws;                                // 8 MB combined hi|lo tiles
  float* wnorm = (float*)(ws + (size_t)8 * 1024 * 1024);                   // 16 KB
  int* idxInt = (int*)(ws + (size_t)8 * 1024 * 1024 + 64 * 1024);          // 128 KB
  float* acc = (float*)(ws + (size_t)8 * 1024 * 1024 + 64 * 1024 + 256 * 1024);

  float* outQ = out + 1;
  float* outIdx = out + 1 + (size_t)N_ROWS * D_DIM;

  vq_prep_kernel<<<K_CODES, 256, 0, stream>>>(weight, wF, wnorm, acc);

  const size_t smem = (size_t)(2 * A_HALVES + 2 * B_HALVES) * sizeof(unsigned short); // 327,680 B
  vq_argmin_kernel<<<N_ROWS / 128, 256, smem, stream>>>(inputs, wF, wnorm, idxInt, outIdx);

  vq_gather_loss_kernel<<<N_ROWS / 2, 256, 0, stream>>>(inputs, weight, idxInt, outQ, acc);
  vq_finalize_kernel<<<1, 1, 0, stream>>>(acc, out);
}
